// HANEncoder_62508954026234
// MI455X (gfx1250) — matrix operand order repaired and checked by reading, not compiled, not run on hardware
//
#include <hip/hip_runtime.h>
#include <hip/hip_bf16.h>

typedef __attribute__((ext_vector_type(16))) __bf16 v16bf;
typedef __attribute__((ext_vector_type(8)))  float  v8f;
typedef int v4i_vs __attribute__((vector_size(16)));   // matches builtin param pointee

#define AS1 __attribute__((address_space(1)))
#define AS3 __attribute__((address_space(3)))

// ---------------------------------------------------------------- utilities
__device__ __forceinline__ unsigned f2ord(float f) {
    unsigned u = __float_as_uint(f);
    return (u & 0x80000000u) ? ~u : (u | 0x80000000u);
}
__device__ __forceinline__ float ord2f(unsigned u) {
    return (u & 0x80000000u) ? __uint_as_float(u & 0x7fffffffu)
                             : __uint_as_float(~u);
}
#define ORD_NEG_INF 0x007FFFFFu   // f2ord(-inf)

__global__ void fill_u32(unsigned* __restrict__ p, unsigned v, long n) {
    long i = (long)blockIdx.x * blockDim.x + threadIdx.x;
    if (i < n) p[i] = v;
}
__global__ void fill_f32(float* __restrict__ p, float v, long n) {
    long i = (long)blockIdx.x * blockDim.x + threadIdx.x;
    if (i < n) p[i] = v;
}

// ------------------------------------------------- WMMA GEMM  Y = X[ridx] @ W
// X: [*, 128] f32 (rows selected by optional ridx), W: [128, NT*16] f32,
// Y: [n, NT*16] f32.  bf16 inputs, f32 accumulate via v_wmma_f32_16x16x32_bf16.
// One wave handles a 16-row M tile and all NT column tiles. 8 waves / block.
// Weight panel staged into LDS with ASYNCcnt-tracked direct-to-LDS loads
// (global_load_async_to_lds_b128), drained with s_wait_asynccnt before the
// workgroup barrier.
template <int NT>
__global__ __launch_bounds__(256) void gemm_bf16_wmma(
    const float* __restrict__ X, const int* __restrict__ ridx,
    const float* __restrict__ W, float* __restrict__ Y, int n) {
    constexpr int NOUT = NT * 16;
    constexpr int K = 128;
    __shared__ float Wl[K * NOUT];

#if __has_builtin(__builtin_amdgcn_global_load_async_to_lds_b128)
    // 16B per async op, LDS written directly (ASYNCcnt), no VGPR staging.
    for (int i = threadIdx.x * 4; i < K * NOUT; i += 256 * 4) {
        __builtin_amdgcn_global_load_async_to_lds_b128(
            (AS1 v4i_vs*)(unsigned long long)(const void*)(W + i),
            (AS3 v4i_vs*)(unsigned)(unsigned long long)(const void*)(Wl + i),
            0, 0);
    }
#if __has_builtin(__builtin_amdgcn_s_wait_asynccnt)
    __builtin_amdgcn_s_wait_asynccnt(0);
#else
    asm volatile("s_wait_asynccnt 0x0" ::: "memory");
#endif
#else
    for (int i = threadIdx.x; i < K * NOUT; i += 256) Wl[i] = W[i];
#endif
    __syncthreads();

    const int lane  = threadIdx.x & 31;
    const int wid   = threadIdx.x >> 5;
    const int wtile = blockIdx.x * 8 + wid;          // 16-row tile id
    if (wtile * 16 >= n) return;                     // wave-uniform: EXEC stays full
    const int half   = lane >> 4;
    const int rlocal = wtile * 16 + (lane & 15);
    const int row    = ridx ? ridx[rlocal] : rlocal;
    const float* xrow = X + (size_t)row * K;
    __builtin_prefetch(xrow, 0, 0);                  // global_prefetch_b8 (speculative)
    __builtin_prefetch(xrow + 64, 0, 0);

    v8f acc[NT];
    v8f zero = {0.f, 0.f, 0.f, 0.f, 0.f, 0.f, 0.f, 0.f};
#pragma unroll
    for (int t = 0; t < NT; ++t) acc[t] = zero;

#pragma unroll
    for (int k0 = 0; k0 < K; k0 += 32) {
        // A fragment: 16x32 bf16. element e -> k = (e>>3)*16 + half*8 + (e&7)
        v16bf a;
        const float* pa0 = xrow + k0 + half * 8;
        const float* pa1 = xrow + k0 + 16 + half * 8;
#pragma unroll
        for (int e = 0; e < 8; ++e) a[e] = (__bf16)pa0[e];
#pragma unroll
        for (int e = 0; e < 8; ++e) a[8 + e] = (__bf16)pa1[e];
#pragma unroll
        for (int t = 0; t < NT; ++t) {
            const int col = t * 16 + (lane & 15);
            // B fragment: 32x16 bf16. element e -> k = (e>>3)*16 + half*8 + (e&7)
            v16bf b;
#pragma unroll
            for (int e = 0; e < 16; ++e)
                b[e] = (__bf16)Wl[(k0 + (e >> 3) * 16 + half * 8 + (e & 7)) * NOUT + col];
            acc[t] = __builtin_amdgcn_wmma_f32_16x16x32_bf16(
                false, a, false, b, (short)0, acc[t], false, false);
        }
    }
    // C/D layout: VGPR j -> row (half*8 + j), col = t*16 + lane%16
#pragma unroll
    for (int t = 0; t < NT; ++t) {
        const int col = t * 16 + (lane & 15);
#pragma unroll
        for (int j = 0; j < 8; ++j) {
            const int r = wtile * 16 + half * 8 + j;
            Y[(size_t)r * NOUT + col] = acc[t][j];
        }
    }
}

// ------------------------------------------------ attention-logit row dots
// adot[i,h] = sum_c Y[i, h*C + c] * avec[h*C + c]
__global__ void rowdot(const float* __restrict__ Yf, const float* __restrict__ avec,
                       float* __restrict__ adot, int n, int Hh, int Cc) {
    int idx = blockIdx.x * blockDim.x + threadIdx.x;
    if (idx >= n * Hh) return;
    int i = idx / Hh, h = idx - i * Hh;
    const float* p  = Yf + (size_t)i * Hh * Cc + h * Cc;
    const float* av = avec + h * Cc;
    float s = 0.f;
    for (int c = 0; c < Cc; ++c) s += p[c] * av[c];
    adot[idx] = s;
}

// ---------------------------------------------------------- edge pass A/B/C
__global__ void edge_logits(const int* __restrict__ ei, const float* __restrict__ asrc,
                            const float* __restrict__ adst, float* __restrict__ elog,
                            unsigned* __restrict__ mmax, int E, int Hh) {
    int idx = blockIdx.x * blockDim.x + threadIdx.x;
    if (idx >= E * Hh) return;
    int e = idx / Hh, h = idx - e * Hh;
    int s = ei[e], d = ei[E + e];
    float v = asrc[s * Hh + h] + adst[d * Hh + h];
    v = v > 0.f ? v : 0.2f * v;          // leaky_relu(0.2)
    elog[idx] = v;
    atomicMax(&mmax[d * Hh + h], f2ord(v));
}

__global__ void finalize_max(unsigned* __restrict__ m, long n) {
    long i = (long)blockIdx.x * blockDim.x + threadIdx.x;
    if (i >= n) return;
    unsigned u = m[i];
    float f = (u == ORD_NEG_INF) ? 0.f : ord2f(u);   // empty segments -> 0
    ((float*)m)[i] = f;
}

__global__ void edge_exp(const int* __restrict__ ei, float* __restrict__ elog,
                         const float* __restrict__ mf, float* __restrict__ den,
                         int E, int Hh) {
    int idx = blockIdx.x * blockDim.x + threadIdx.x;
    if (idx >= E * Hh) return;
    int e = idx / Hh, h = idx - e * Hh;
    int d = ei[E + e];
    float p = __expf(elog[idx] - mf[d * Hh + h]);
    elog[idx] = p;
    atomicAdd(&den[d * Hh + h], p);
}

// one thread per (edge, 4 features): b128 gather + 4 f32 atomics
__global__ void edge_scatter(const int* __restrict__ ei, const float* __restrict__ elog,
                             const float* __restrict__ den, const float* __restrict__ hs,
                             float* __restrict__ o, int E, int Hh, int Cc) {
    const int F = Hh * Cc;
    const int F4 = F >> 2;
    long idx = (long)blockIdx.x * blockDim.x + threadIdx.x;
    if (idx >= (long)E * F4) return;
    int e  = (int)(idx / F4);
    int f  = (int)(idx - (long)e * F4) * 4;
    int h  = f / Cc;                 // Cc % 4 == 0 -> chunk never straddles heads
    int s = ei[e], d = ei[E + e];
    float alpha = elog[e * Hh + h] / (den[d * Hh + h] + 1e-16f);
    const float4 hv = *(const float4*)(hs + (size_t)s * F + f);
    float* op = o + (size_t)d * F + f;
    atomicAdd(op + 0, hv.x * alpha);
    atomicAdd(op + 1, hv.y * alpha);
    atomicAdd(op + 2, hv.z * alpha);
    atomicAdd(op + 3, hv.w * alpha);
}

// ----------------------------------- sem-gate + bias + LayerNorm (+ELU) epilogue
// one wave32 per node row
__global__ void node_epilogue(const float* __restrict__ o, const float* __restrict__ bias,
                              const float* __restrict__ sem, const float* __restrict__ g,
                              const float* __restrict__ bta, float* __restrict__ out,
                              int n, int F, int elu) {
    int wid  = (int)(((long)blockIdx.x * blockDim.x + threadIdx.x) >> 5);
    int lane = threadIdx.x & 31;
    if (wid >= n) return;
    float w  = 1.f / (1.f + __expf(-sem[0]));
    int   FP = F >> 5;                     // 4 (F=128) or 2 (F=64)
    float vals[4];
    float sum = 0.f;
    for (int i = 0; i < FP; ++i) {
        int f = lane * FP + i;
        float v = w * (o[(size_t)wid * F + f] + bias[f]);
        vals[i] = v;
        sum += v;
    }
    for (int m = 16; m >= 1; m >>= 1) sum += __shfl_xor(sum, m, 32);
    float mu = sum / F;
    float var = 0.f;
    for (int i = 0; i < FP; ++i) { float d = vals[i] - mu; var += d * d; }
    for (int m = 16; m >= 1; m >>= 1) var += __shfl_xor(var, m, 32);
    float inv = rsqrtf(var / F + 1e-5f);
    for (int i = 0; i < FP; ++i) {
        int f = lane * FP + i;
        float y = (vals[i] - mu) * inv * g[f] + bta[f];
        if (elu) y = y > 0.f ? y : __expf(y) - 1.f;
        out[(size_t)wid * F + f] = y;
    }
}

// --------------------------------------------------------------- host side
static inline int cdiv(long a, int b) { return (int)((a + b - 1) / b); }

struct GatBufs {
    float* hs; float* hd; float* asrc; float* adst;
    unsigned* mmax; float* den; float* elog;
};

static void run_gat(const float* Xs, const int* xsIdx, const float* Xd, const int* xdIdx,
                    const int* ei, const float* Ws, const float* Wd,
                    const float* avs, const float* avd, const float* bias,
                    const float* sem, const float* lng, const float* lnb,
                    float* dest, int n, int E, int Hh, int Cc, int elu,
                    const GatBufs& B, hipStream_t stream) {
    const int F = Hh * Cc;
    dim3 blk(256);
    int gblocks = cdiv(n / 16, 8);
    if (F == 128) {
        gemm_bf16_wmma<8><<<gblocks, blk, 0, stream>>>(Xs, xsIdx, Ws, B.hs, n);
        gemm_bf16_wmma<8><<<gblocks, blk, 0, stream>>>(Xd, xdIdx, Wd, B.hd, n);
    } else {
        gemm_bf16_wmma<4><<<gblocks, blk, 0, stream>>>(Xs, xsIdx, Ws, B.hs, n);
        gemm_bf16_wmma<4><<<gblocks, blk, 0, stream>>>(Xd, xdIdx, Wd, B.hd, n);
    }
    rowdot<<<cdiv((long)n * Hh, 256), blk, 0, stream>>>(B.hs, avs, B.asrc, n, Hh, Cc);
    rowdot<<<cdiv((long)n * Hh, 256), blk, 0, stream>>>(B.hd, avd, B.adst, n, Hh, Cc);
    // hd is dead after adst -> reuse it as the aggregation buffer
    float* o = B.hd;
    fill_u32<<<cdiv((long)n * Hh, 256), blk, 0, stream>>>(B.mmax, ORD_NEG_INF, (long)n * Hh);
    fill_f32<<<cdiv((long)n * Hh, 256), blk, 0, stream>>>(B.den, 0.f, (long)n * Hh);
    fill_f32<<<cdiv((long)n * F, 256), blk, 0, stream>>>(o, 0.f, (long)n * F);
    edge_logits<<<cdiv((long)E * Hh, 256), blk, 0, stream>>>(ei, B.asrc, B.adst, B.elog, B.mmax, E, Hh);
    finalize_max<<<cdiv((long)n * Hh, 256), blk, 0, stream>>>(B.mmax, (long)n * Hh);
    edge_exp<<<cdiv((long)E * Hh, 256), blk, 0, stream>>>(ei, B.elog, (const float*)B.mmax, B.den, E, Hh);
    edge_scatter<<<cdiv((long)E * (F >> 2), 256), blk, 0, stream>>>(ei, B.elog, B.den, B.hs, o, E, Hh, Cc);
    node_epilogue<<<cdiv((long)n * 32, 256), blk, 0, stream>>>(o, bias, sem, lng, lnb, dest, n, F, elu);
}

extern "C" void kernel_launch(void* const* d_in, const int* in_sizes, int n_in,
                              void* d_out, int out_size, void* d_ws, size_t ws_size,
                              hipStream_t stream) {
    const int N = in_sizes[0];
    const int E = in_sizes[2] / 2;
    const int HID = 128, OUT = 64, H = 4, C = 32;

    const int*   x_drug   = (const int*)d_in[0];
    const int*   x_dis    = (const int*)d_in[1];
    const int*   ei_dt    = (const int*)d_in[2];
    const int*   ei_td    = (const int*)d_in[3];
    const float* emb_drug = (const float*)d_in[4];
    const float* emb_dis  = (const float*)d_in[5];
    // dt: 6..15, td: 16..25
    const float* Ws0_dt = (const float*)d_in[6];
    const float* Wd0_dt = (const float*)d_in[7];
    const float* as0_dt = (const float*)d_in[8];
    const float* ad0_dt = (const float*)d_in[9];
    const float* b0_dt  = (const float*)d_in[10];
    const float* Ws1_dt = (const float*)d_in[11];
    const float* Wd1_dt = (const float*)d_in[12];
    const float* as1_dt = (const float*)d_in[13];
    const float* ad1_dt = (const float*)d_in[14];
    const float* b1_dt  = (const float*)d_in[15];
    const float* Ws0_td = (const float*)d_in[16];
    const float* Wd0_td = (const float*)d_in[17];
    const float* as0_td = (const float*)d_in[18];
    const float* ad0_td = (const float*)d_in[19];
    const float* b0_td  = (const float*)d_in[20];
    const float* Ws1_td = (const float*)d_in[21];
    const float* Wd1_td = (const float*)d_in[22];
    const float* as1_td = (const float*)d_in[23];
    const float* ad1_td = (const float*)d_in[24];
    const float* b1_td  = (const float*)d_in[25];
    const float* ln0_drug_g = (const float*)d_in[26];
    const float* ln0_drug_b = (const float*)d_in[27];
    const float* ln1_drug_g = (const float*)d_in[28];
    const float* ln1_drug_b = (const float*)d_in[29];
    const float* ln0_dis_g  = (const float*)d_in[30];
    const float* ln0_dis_b  = (const float*)d_in[31];
    const float* ln1_dis_g  = (const float*)d_in[32];
    const float* ln1_dis_b  = (const float*)d_in[33];
    const float* sem_dt = (const float*)d_in[34];
    const float* sem_td = (const float*)d_in[35];

    float* out = (float*)d_out;
    float* ws  = (float*)d_ws;
    size_t off = 0;
    auto alloc = [&](size_t nf) { float* p = ws + off; off += nf; return p; };

    float* A_drug = alloc((size_t)N * HID);   // layer-0 activations
    float* A_dis  = alloc((size_t)N * HID);
    GatBufs B;
    B.hs   = alloc((size_t)N * HID);
    B.hd   = alloc((size_t)N * HID);          // doubles as aggregation buffer
    B.asrc = alloc((size_t)N * H);
    B.adst = alloc((size_t)N * H);
    B.mmax = (unsigned*)alloc((size_t)N * H);
    B.den  = alloc((size_t)N * H);
    B.elog = alloc((size_t)E * H);
    (void)n_in; (void)out_size; (void)ws_size;

    // ---- layer 0 (H=4, C=32, concat, LN+ELU), indices gather original emb
    run_gat(emb_drug, x_drug, emb_dis, x_dis, ei_dt,
            Ws0_dt, Wd0_dt, as0_dt, ad0_dt, b0_dt, sem_dt,
            ln0_dis_g, ln0_dis_b, A_dis, N, E, H, C, 1, B, stream);
    run_gat(emb_dis, x_dis, emb_drug, x_drug, ei_td,
            Ws0_td, Wd0_td, as0_td, ad0_td, b0_td, sem_td,
            ln0_drug_g, ln0_drug_b, A_drug, N, E, H, C, 1, B, stream);

    // ---- layer 1 (H=1, C=64, mean==identity, LN only) -> d_out
    run_gat(A_drug, nullptr, A_dis, nullptr, ei_dt,
            Ws1_dt, Wd1_dt, as1_dt, ad1_dt, b1_dt, sem_dt,
            ln1_dis_g, ln1_dis_b, out + (size_t)N * OUT, N, E, 1, OUT, 0, B, stream);
    run_gat(A_dis, nullptr, A_drug, nullptr, ei_td,
            Ws1_td, Wd1_td, as1_td, ad1_td, b1_td, sem_td,
            ln1_drug_g, ln1_drug_b, out, N, E, 1, OUT, 0, B, stream);
}
